// PointPillarsBackbone_10874857193826
// MI455X (gfx1250) — matrix operand order repaired and checked by reading, not compiled, not run on hardware
//
#include <hip/hip_runtime.h>
#include <hip/hip_bf16.h>

// ---------------------------------------------------------------------------
// Types & helpers
// ---------------------------------------------------------------------------
typedef __bf16 bf16_t;
typedef bf16_t bf16x16 __attribute__((ext_vector_type(16)));
typedef bf16_t bf16x8  __attribute__((ext_vector_type(8)));
typedef float  floatx8 __attribute__((ext_vector_type(8)));

// native converts (v_cvt_* on gfx1250)
__device__ __forceinline__ bf16_t f2bf(float f) { return (bf16_t)f; }
__device__ __forceinline__ float  bf2f(bf16_t h) { return (float)h; }

#define CAT16(lo, hi) \
    __builtin_shufflevector((lo), (hi), 0,1,2,3,4,5,6,7,8,9,10,11,12,13,14,15)

#define WMMA_BF16(A, Bm, C) \
    __builtin_amdgcn_wmma_f32_16x16x32_bf16(false, (A), false, (Bm), (short)0, (C), false, false)

// Problem constants (from reference)
#define PB   4
#define PP   12000
#define NPT  100
#define H0   108
#define W0   124
#define NEG_INF (-3.402823466e38f)

// A-matrix (16x32 bf16) per-lane K index for element i (ISA 7.12.2):
// two contiguous 8-element runs: [half*8, +8) and [16+half*8, +8)
__device__ __forceinline__ int a_kidx(int i, int half) {
    return (i & 7) + ((i >> 3) << 4) + (half << 3);
}

// ---------------------------------------------------------------------------
// Kernel 0: zero the mean accumulators (d_ws is poisoned by harness)
// ---------------------------------------------------------------------------
__global__ void pp_init_kernel(float* __restrict__ sums, int* __restrict__ cnt) {
    int i = threadIdx.x;
    if (i < PB * 64) sums[i] = 0.0f;
    if (i < PB)      cnt[i]  = 0;
}

// ---------------------------------------------------------------------------
// Weight prep: HWIO fp32 [ks,ks,Cin,Cout] -> bf16 [CoutPad][Ktot], k=tap*Cin+ci
// (rows co >= Cout are zero so the conv kernel never needs load masks)
// ---------------------------------------------------------------------------
__global__ void pp_wprep_kernel(const float* __restrict__ wk, bf16_t* __restrict__ wT,
                                int Cout, int CoutPad, int Ktot) {
    int idx = blockIdx.x * 256 + threadIdx.x;
    if (idx >= CoutPad * Ktot) return;
    int co = idx / Ktot;
    int k  = idx - co * Ktot;
    float v = (co < Cout) ? wk[(long long)k * Cout + co] : 0.0f;
    wT[idx] = f2bf(v);
}

// PFN weights: wT1 [64][32] (K padded 9->32 with zeros), wT2 [64][64]
__global__ void pp_pfnprep_kernel(const float* __restrict__ w1,
                                  const float* __restrict__ w2,
                                  bf16_t* __restrict__ wt1,
                                  bf16_t* __restrict__ wt2) {
    int idx = blockIdx.x * 256 + threadIdx.x;
    if (idx < 64 * 32) {
        int co = idx >> 5, k = idx & 31;
        wt1[idx] = f2bf((k < 9) ? w1[k * 64 + co] : 0.0f);
    }
    if (idx < 64 * 64) {
        int co = idx >> 6, k = idx & 63;
        wt2[idx] = f2bf(w2[k * 64 + co]);
    }
}

// ---------------------------------------------------------------------------
// Kernel 1: Pillar Feature Net (9->64->64 MLP, folded BN, ReLU) + masked
//           point-max + masked pillar-mean accumulation.
// One block = one pillar; 4 waves, each owns 16 of the 64 output channels.
// ---------------------------------------------------------------------------
__global__ __launch_bounds__(128) void pp_pfn_kernel(
    const float* __restrict__ pillars,     // [B,P,N,9]
    const bf16_t* __restrict__ wt1,        // [64][32] bf16 (co-major)
    const bf16_t* __restrict__ wt2,        // [64][64] bf16 (co-major)
    const float* __restrict__ b1, const float* __restrict__ s1, const float* __restrict__ t1,
    const float* __restrict__ b2, const float* __restrict__ s2, const float* __restrict__ t2,
    const int*   __restrict__ num_points,  // [B,P]
    float* __restrict__ mean_sum,          // [B,64]
    int*   __restrict__ valid_cnt)         // [B]
{
    const int pillar = blockIdx.x;
    const int b      = pillar / PP;
    const int np     = num_points[pillar];

    const int wave = threadIdx.x >> 5;
    const int lane = threadIdx.x & 31;
    const int half = lane >> 4;
    const int l16  = lane & 15;
    const int ncol = wave * 16 + l16;             // output channel

    __shared__ __align__(16) bf16_t hbuf[16][72]; // 144B row stride (16B aligned)

    // ---- weight (B-matrix) fragments: b128 loads, k runs [half*8,+8),[16+half*8,+8)
    bf16x16 B1, B2lo, B2hi;
    {
        const bf16_t* r1 = wt1 + ncol * 32 + 8 * half;
        B1 = CAT16(*(const bf16x8*)r1, *(const bf16x8*)(r1 + 16));
        const bf16_t* r2 = wt2 + ncol * 64 + 8 * half;
        B2lo = CAT16(*(const bf16x8*)r2,        *(const bf16x8*)(r2 + 16));
        B2hi = CAT16(*(const bf16x8*)(r2 + 32), *(const bf16x8*)(r2 + 48));
    }
    const float bb1 = b1[ncol], ss1 = s1[ncol], tt1 = t1[ncol];
    const float bb2 = b2[ncol], ss2 = s2[ncol], tt2 = t2[ncol];

    const float* pbase = pillars + (long long)pillar * (NPT * 9);

    float colmax = NEG_INF;

    for (int mt = 0; mt < 7; ++mt) {
        // ---- layer 1: A = points[16 x 9->32] ------------------------------
        // Per-lane live fragment slots: lanes 0-15 need k=0..7 (8 contiguous
        // floats), lanes 16-31 only k=8. Clamp the row so loads stay in
        // bounds; mask the row validity with selects (no per-element branches).
        const int row  = mt * 16 + l16;
        const bool rv  = (row < NPT);
        const float* rp = pbase + (rv ? row : (NPT - 1)) * 9;

        float r[8];
        if (half == 0) {
#pragma unroll
            for (int j = 0; j < 8; ++j) r[j] = rp[j];      // k = 0..7
        } else {
            r[0] = rp[8];                                   // k = 8
#pragma unroll
            for (int j = 1; j < 8; ++j) r[j] = 0.0f;        // k = 9..15 pad
        }
        bf16x16 A;
#pragma unroll
        for (int j = 0; j < 8; ++j) A[j] = f2bf(rv ? r[j] : 0.0f);
#pragma unroll
        for (int j = 8; j < 16; ++j) A[j] = f2bf(0.0f);     // k >= 16 pad

        floatx8 c = {};
        c = WMMA_BF16(A, B1, c);

        __syncthreads();
#pragma unroll
        for (int g = 0; g < 8; ++g) {
            float h = (c[g] + bb1) * ss1 + tt1;
            h = h > 0.0f ? h : 0.0f;
            hbuf[g + 8 * half][ncol] = f2bf(h);
        }
        __syncthreads();

        // ---- layer 2: A from LDS via 4 x ds_load_b128 ---------------------
        const bf16_t* hr = &hbuf[l16][0];
        bf16x16 A2lo = CAT16(*(const bf16x8*)(hr + half * 8),
                             *(const bf16x8*)(hr + 16 + half * 8));
        bf16x16 A2hi = CAT16(*(const bf16x8*)(hr + 32 + half * 8),
                             *(const bf16x8*)(hr + 48 + half * 8));
        floatx8 c2 = {};
        c2 = WMMA_BF16(A2lo, B2lo, c2);
        c2 = WMMA_BF16(A2hi, B2hi, c2);

#pragma unroll
        for (int g = 0; g < 8; ++g) {
            float h = (c2[g] + bb2) * ss2 + tt2;
            h = h > 0.0f ? h : 0.0f;
            int pt = mt * 16 + g + 8 * half;
            if (pt < np) colmax = fmaxf(colmax, h);
        }
    }

    float other = __shfl_xor(colmax, 16, 32);
    colmax = fmaxf(colmax, other);

    if (np > 0) {
        if (half == 0) atomicAdd(&mean_sum[b * 64 + ncol], colmax);
        if (threadIdx.x == 0) atomicAdd(&valid_cnt[b], 1);
    }
}

// ---------------------------------------------------------------------------
// Kernel 2: broadcast mean[B,64] to bf16 NHWC pseudo-image [B,H0,W0,64]
// ---------------------------------------------------------------------------
__global__ void pp_fill_kernel(const float* __restrict__ mean_sum,
                               const int*   __restrict__ valid_cnt,
                               bf16_t* __restrict__ out) {
    const int HW = H0 * W0;
    int tid = blockIdx.x * blockDim.x + threadIdx.x;
    int total = PB * HW * 8;                 // 8 channels per thread
    if (tid >= total) return;
    int cg = (tid & 7) * 8;
    int p  = tid >> 3;                       // b*HW + pix
    int b  = p / HW;
    int cnt = valid_cnt[b]; if (cnt < 1) cnt = 1;
    float inv = 1.0f / (float)cnt;
    bf16x8 v;
#pragma unroll
    for (int j = 0; j < 8; ++j) v[j] = f2bf(mean_sum[b * 64 + cg + j] * inv);
    *(bf16x8*)(out + (long long)p * 64 + cg) = v;
}

// ---------------------------------------------------------------------------
// Kernel 3: implicit-GEMM conv via bf16 WMMA, NHWC activations.
//   D[pix, co] = sum_k im2col[pix, k] * Wt[co, k],  k = tap*Cin + ci
// A = im2col (M=16 pixels), B = weights (N=16 couts). Each wave does two
// pixel tiles (reuses the B fragment). All fragment loads are b128s.
// ---------------------------------------------------------------------------
template <bool OUT_F32>
__global__ __launch_bounds__(128) void pp_conv_kernel(
    const bf16_t* __restrict__ in,    // [B,H,W,Cin] bf16
    void* __restrict__ outp,          // [B,H,W,Cout] bf16  /  [B,Cout,H,W] f32
    const bf16_t* __restrict__ wT,    // [CoutPad][Ktot] bf16
    const float* __restrict__ bias,   // [Cout]
    const float* __restrict__ scale,  // [Cout] or nullptr
    const float* __restrict__ shift,  // [Cout] or nullptr
    int Cin, int Cout, int H, int W, int ks, int relu)
{
    const int wave = threadIdx.x >> 5;
    const int lane = threadIdx.x & 31;
    const int half = lane >> 4;
    const int l16  = lane & 15;

    const int HW    = H * W;
    const int tile0 = (blockIdx.x * 4 + wave) * 2;   // two pixel tiles / wave
    if (tile0 * 16 >= HW) return;                    // wave-uniform exit

    const int bb = blockIdx.z;
    const int co = blockIdx.y * 16 + l16;            // B column (cout)
    const int pad  = ks >> 1;
    const int cmask = Cin - 1;
    const int csh   = 31 - __clz(Cin);               // Cin is a power of two
    const int Ktot   = ks * ks * Cin;
    const int nchunk = Ktot >> 5;

    const bf16_t* inb  = in + (long long)bb * HW * Cin;
    const bf16_t* wrow = wT + (long long)co * Ktot;

    int  py[2], px[2];
    bool pok[2];
#pragma unroll
    for (int t = 0; t < 2; ++t) {
        int pix = (tile0 + t) * 16 + l16;            // A row (M) for this lane
        pok[t] = pix < HW;
        int pp = pok[t] ? pix : 0;
        py[t] = pp / W;
        px[t] = pp % W;
    }

    floatx8 acc[2] = {{}, {}};
    for (int kc = 0; kc < nchunk; ++kc) {
        const int kbase = kc << 5;
        // B fragment: k runs [kbase+8*half,+8) and [kbase+16+8*half,+8)  (2 x b128)
        const bf16_t* wp = wrow + kbase + 8 * half;
        bf16x16 Bm = CAT16(*(const bf16x8*)wp, *(const bf16x8*)(wp + 16));

        // one 3x3 tap per 32-K chunk (Cin >= 32)
        const int tap = kbase >> csh;
        const int dy  = tap / ks - pad;
        const int dx  = tap % ks - pad;
        const int ci0 = (kbase & cmask) + half * 8;  // A run0 channel start

#pragma unroll
        for (int t = 0; t < 2; ++t) {
            bf16x8 a0 = {}, a1 = {};
            int y = py[t] + dy, x = px[t] + dx;
            if (pok[t] && y >= 0 && y < H && x >= 0 && x < W) {
                const bf16_t* p = inb + ((long long)(y * W + x)) * Cin + ci0;
                a0 = *(const bf16x8*)p;              // k runs: [half*8, +8)
                a1 = *(const bf16x8*)(p + 16);       //          [16+half*8, +8)
            }
            bf16x16 A = CAT16(a0, a1);
            acc[t] = WMMA_BF16(A, Bm, acc[t]);
        }
    }

    // epilogue: per-lane single cout -> one set of channel params
    const bool  cok = (co < Cout);
    const float bbv = cok ? bias[co] : 0.0f;
    const float ssv = (cok && scale) ? scale[co] : 1.0f;
    const float ttv = (cok && scale) ? shift[co] : 0.0f;

#pragma unroll
    for (int t = 0; t < 2; ++t) {
#pragma unroll
        for (int g = 0; g < 8; ++g) {
            int pix = (tile0 + t) * 16 + g + 8 * half;   // C row (M) = pixel
            if (pix >= HW || !cok) continue;
            float v = acc[t][g] + bbv;
            v = v * ssv + ttv;
            if (relu && v < 0.0f) v = 0.0f;
            if (OUT_F32)   // final head: NCHW fp32 to d_out
                ((float*)outp)[((long long)bb * Cout + co) * HW + pix] = v;
            else           // NHWC bf16 intermediate
                ((bf16_t*)outp)[((long long)bb * HW + pix) * Cout + co] = f2bf(v);
        }
    }
}

// ---------------------------------------------------------------------------
// Kernel 4: 2x2 max-pool stride 2, NHWC bf16, 8 channels / thread
// ---------------------------------------------------------------------------
__global__ void pp_pool_kernel(const bf16_t* __restrict__ in,
                               bf16_t* __restrict__ out,
                               int C, int H, int W) {
    const int H2 = H >> 1, W2 = W >> 1;
    const int cg8 = C >> 3;
    int tid = blockIdx.x * blockDim.x + threadIdx.x;
    int total = PB * H2 * W2 * cg8;
    if (tid >= total) return;
    int cg = (tid % cg8) * 8;
    int p  = tid / cg8;
    int x  = p % W2;
    int r  = p / W2;
    int y  = r % H2;
    int b  = r / H2;
    const bf16_t* base = in + (((long long)b * H + 2 * y) * W + 2 * x) * C + cg;
    bf16x8 v0 = *(const bf16x8*)base;
    bf16x8 v1 = *(const bf16x8*)(base + C);
    bf16x8 v2 = *(const bf16x8*)(base + (long long)W * C);
    bf16x8 v3 = *(const bf16x8*)(base + (long long)(W + 1) * C);
    bf16x8 rv;
#pragma unroll
    for (int j = 0; j < 8; ++j)
        rv[j] = f2bf(fmaxf(fmaxf(bf2f(v0[j]), bf2f(v1[j])),
                           fmaxf(bf2f(v2[j]), bf2f(v3[j]))));
    *(bf16x8*)(out + (((long long)b * H2 + y) * W2 + x) * C + cg) = rv;
}

// ---------------------------------------------------------------------------
// Host-side launcher
// ---------------------------------------------------------------------------
static void launch_conv(hipStream_t stream, const bf16_t* in, void* out, bool out_f32,
                        const bf16_t* wT, const float* bias,
                        const float* scale, const float* shift,
                        int Cin, int Cout, int CoutPad, int H, int W, int ks, int relu) {
    int HW = H * W;
    int ntiles = (HW + 15) / 16;
    dim3 grid((ntiles + 7) / 8, CoutPad / 16, PB);
    if (out_f32)
        pp_conv_kernel<true><<<grid, 128, 0, stream>>>(in, out, wT, bias, scale, shift,
                                                       Cin, Cout, H, W, ks, relu);
    else
        pp_conv_kernel<false><<<grid, 128, 0, stream>>>(in, out, wT, bias, scale, shift,
                                                        Cin, Cout, H, W, ks, relu);
}

extern "C" void kernel_launch(void* const* d_in, const int* in_sizes, int n_in,
                              void* d_out, int out_size, void* d_ws, size_t ws_size,
                              hipStream_t stream) {
    (void)in_sizes; (void)n_in; (void)out_size; (void)ws_size;

    const float* pillars = (const float*)d_in[0];
    const float* w1 = (const float*)d_in[1],  *b1 = (const float*)d_in[2];
    const float* s1 = (const float*)d_in[3],  *t1 = (const float*)d_in[4];
    const float* w2 = (const float*)d_in[5],  *b2 = (const float*)d_in[6];
    const float* s2 = (const float*)d_in[7],  *t2 = (const float*)d_in[8];
    const float* ck[6], *cb[6], *cs[6], *ct[6];
    for (int i = 0; i < 6; ++i) {
        ck[i] = (const float*)d_in[9 + 4 * i];
        cb[i] = (const float*)d_in[10 + 4 * i];
        cs[i] = (const float*)d_in[11 + 4 * i];
        ct[i] = (const float*)d_in[12 + 4 * i];
    }
    const float* hk1 = (const float*)d_in[33], *hb1 = (const float*)d_in[34];
    const float* hk2 = (const float*)d_in[35], *hb2 = (const float*)d_in[36];
    const int* num_points = (const int*)d_in[37];

    // ---- workspace layout -------------------------------------------------
    float*  mean_sum = (float*)d_ws;                          // 256 f32
    int*    valid    = (int*)((char*)d_ws + 1024);            // 4 i32
    bf16_t* wt1      = (bf16_t*)((char*)d_ws + 2048);         // 64*32
    bf16_t* wt2      = wt1 + 64 * 32;                         // 64*64
    bf16_t* wbase    = (bf16_t*)((char*)d_ws + 16384);

    // conv weight slabs
    const int convCin [8] = { 64,  64,  64, 128, 128, 256, 256, 256};
    const int convCout[8] = { 64,  64, 128, 128, 256, 256, 256,  10};
    const int convKs  [8] = {  3,   3,   3,   3,   3,   3,   3,   1};
    const float* convW[8] = {ck[0], ck[1], ck[2], ck[3], ck[4], ck[5], hk1, hk2};
    bf16_t* wTs[8];
    size_t  wo = 0;
    int     cpads[8], ktots[8];
    for (int i = 0; i < 8; ++i) {
        int ktot = convKs[i] * convKs[i] * convCin[i];
        int cpad = (convCout[i] + 15) & ~15;
        wTs[i] = wbase + wo;
        wo += (size_t)cpad * ktot;
        cpads[i] = cpad; ktots[i] = ktot;
    }
    bf16_t* bufA = wbase + ((wo + 255) & ~(size_t)255);
    const size_t bufElems = (size_t)PB * 64 * H0 * W0;        // largest tensor
    bf16_t* bufB = bufA + bufElems;

    // ---- weight prep (bf16, transposed) -----------------------------------
    pp_pfnprep_kernel<<<(64 * 64 + 255) / 256, 256, 0, stream>>>(w1, w2, wt1, wt2);
    for (int i = 0; i < 8; ++i) {
        int total = cpads[i] * ktots[i];
        pp_wprep_kernel<<<(total + 255) / 256, 256, 0, stream>>>(
            convW[i], wTs[i], convCout[i], cpads[i], ktots[i]);
    }

    // ---- PFN: WMMA MLP + masked max + mean --------------------------------
    pp_init_kernel<<<1, 256, 0, stream>>>(mean_sum, valid);
    pp_pfn_kernel<<<dim3(PB * PP), 128, 0, stream>>>(
        pillars, wt1, wt2, b1, s1, t1, b2, s2, t2, num_points, mean_sum, valid);

    // ---- broadcast to NHWC pseudo-image [4,108,124,64] --------------------
    {
        int total = PB * H0 * W0 * 8;
        pp_fill_kernel<<<(total + 255) / 256, 256, 0, stream>>>(mean_sum, valid, bufA);
    }

    // ---- backbone ----------------------------------------------------------
    launch_conv(stream, bufA, bufB, false, wTs[0], cb[0], cs[0], ct[0], 64, 64, 64, H0, W0, 3, 1);
    launch_conv(stream, bufB, bufA, false, wTs[1], cb[1], cs[1], ct[1], 64, 64, 64, H0, W0, 3, 1);
    {
        int total = PB * (H0 / 2) * (W0 / 2) * (64 / 8);
        pp_pool_kernel<<<(total + 255) / 256, 256, 0, stream>>>(bufA, bufB, 64, H0, W0);
    }
    const int H1 = H0 / 2, W1 = W0 / 2;        // 54 x 62
    launch_conv(stream, bufB, bufA, false, wTs[2], cb[2], cs[2], ct[2], 64, 128, 128, H1, W1, 3, 1);
    launch_conv(stream, bufA, bufB, false, wTs[3], cb[3], cs[3], ct[3], 128, 128, 128, H1, W1, 3, 1);
    {
        int total = PB * (H1 / 2) * (W1 / 2) * (128 / 8);
        pp_pool_kernel<<<(total + 255) / 256, 256, 0, stream>>>(bufB, bufA, 128, H1, W1);
    }
    const int H2d = H1 / 2, W2d = W1 / 2;      // 27 x 31
    launch_conv(stream, bufA, bufB, false, wTs[4], cb[4], cs[4], ct[4], 128, 256, 256, H2d, W2d, 3, 1);
    launch_conv(stream, bufB, bufA, false, wTs[5], cb[5], cs[5], ct[5], 256, 256, 256, H2d, W2d, 3, 1);

    // ---- head: conv3x3+ReLU (no BN), conv1x1 -> fp32 NCHW output ----------
    launch_conv(stream, bufA, bufB, false, wTs[6], hb1, nullptr, nullptr, 256, 256, 256, H2d, W2d, 3, 1);
    launch_conv(stream, bufB, d_out, true,  wTs[7], hb2, nullptr, nullptr, 256, 10, 16, H2d, W2d, 1, 0);
}
